// VariationalAutoencoder_65034394796406
// MI455X (gfx1250) — matrix operand order repaired and checked by reading, not compiled, not run on hardware
//
#include <hip/hip_runtime.h>
#include <cstdint>
#include <cstddef>

// ---------- types ----------
typedef unsigned short u16;
typedef __attribute__((ext_vector_type(8)))  u16    u16x8;
typedef __attribute__((ext_vector_type(16))) u16    u16x16;
typedef __attribute__((ext_vector_type(16))) __bf16 v16bf;
typedef __attribute__((ext_vector_type(8)))  float  v8f;
typedef __attribute__((ext_vector_type(4)))  unsigned int u32x4;
typedef __attribute__((ext_vector_type(8)))  int    i32x8;
typedef __attribute__((ext_vector_type(4)))  int    i32x4;

#define DEV static __device__ __forceinline__

DEV u16 f2bf(float f) {
  union { float f; unsigned u; } v; v.f = f;
  unsigned r = v.u + 0x7fffu + ((v.u >> 16) & 1u);
  return (u16)(r >> 16);
}
DEV v16bf mkfrag(u16x8 lo, u16x8 hi) {
  u16x16 v = __builtin_shufflevector(lo, hi, 0,1,2,3,4,5,6,7,8,9,10,11,12,13,14,15);
  return __builtin_bit_cast(v16bf, v);
}
DEV v8f wmma_bf16(u16x8 a0, u16x8 a1, u16x8 b0, u16x8 b1, v8f c) {
  return __builtin_amdgcn_wmma_f32_16x16x32_bf16(
      false, mkfrag(a0, a1), false, mkfrag(b0, b1), (short)0, c, false, false);
}
DEV float sigf(float x)      { return 1.f / (1.f + __expf(-x)); }
DEV float softplusf(float x) { return x > 20.f ? x : log1pf(__expf(x)); }

// dims: B=512 T=256 P=61 H=256 G=1024 LAT=512 DEC=32 TB=16 NPB=16
// encoder combined K = 64(x pad) + 256(h) = 320
// decoder combined K = 32(emb) + 64(teacher pad) + 32(h) = 128
// conductor combined K = 32(zdec) + 32(h) = 64

// ---------- pack / convert kernels ----------
__global__ void k_convert_x(const float* __restrict__ x, u16* __restrict__ xbf) {
  int i = blockIdx.x * 256 + threadIdx.x;          // over B*T*64
  if (i >= 512 * 256 * 64) return;
  int k = i & 63; int rt = i >> 6;                 // rt = b*256+t
  xbf[i] = (k < 61) ? f2bf(x[(size_t)rt * 61 + k]) : (u16)0;
}
__global__ void k_f32_to_bf16(const float* __restrict__ s, u16* __restrict__ d, int n) {
  int i = blockIdx.x * 256 + threadIdx.x;
  if (i < n) d[i] = f2bf(s[i]);
}
__global__ void k_pack_enc(const float* __restrict__ Wih, const float* __restrict__ Whh,
                           u16* __restrict__ W) {
  int i = blockIdx.x * 256 + threadIdx.x;          // 1024*320
  if (i >= 1024 * 320) return;
  int k = i % 320, n = i / 320;
  float v = 0.f;
  if (k < 64) { if (k < 61) v = Wih[n * 61 + k]; }
  else v = Whh[n * 256 + (k - 64)];
  W[i] = f2bf(v);
}
__global__ void k_pack_con(const float* __restrict__ Wih, const float* __restrict__ Whh,
                           u16* __restrict__ W) {
  int i = blockIdx.x * 256 + threadIdx.x;          // 128*64
  if (i >= 128 * 64) return;
  int k = i & 63, n = i >> 6;
  float v = (k < 32) ? Wih[n * 32 + k] : Whh[n * 32 + (k - 32)];
  W[i] = f2bf(v);
}
__global__ void k_pack_dec(const float* __restrict__ Wih, const float* __restrict__ Whh,
                           u16* __restrict__ W) {
  int i = blockIdx.x * 256 + threadIdx.x;          // 128*128
  if (i >= 128 * 128) return;
  int k = i & 127, n = i >> 7;
  float v;
  if (k < 96) v = (k < 93) ? Wih[n * 93 + k] : 0.f;
  else        v = Whh[n * 32 + (k - 96)];
  W[i] = f2bf(v);
}
__global__ void k_pack_lin(const float* __restrict__ Wlin, u16* __restrict__ W) {
  int i = blockIdx.x * 256 + threadIdx.x;          // 64*32
  if (i >= 64 * 32) return;
  int k = i & 31, n = i >> 5;
  W[i] = (n < 61) ? f2bf(Wlin[n * 32 + k]) : (u16)0;
}

// ---------- bidirectional encoder LSTM ----------
// grid (32 m-blocks, 2 dirs), 1024 threads = 32 waves (full WGP): each wave only
// 2 N-tiles x 10 K-WMMAs per step -> 4x shorter serial critical path than 8 waves.
// Per-step x-tile (16 rows x 64 bf16) staged into LDS by the Tensor Data Mover.
__global__ __launch_bounds__(1024) void k_encoder(
    const u16* __restrict__ xbf, const u16* __restrict__ Wcat_f,
    const u16* __restrict__ Wcat_b, const float* __restrict__ bias_f,
    const float* __restrict__ bias_b, u16* __restrict__ ybf) {
  __shared__ float sG[16 * 1024];   // gates, 64KB
  __shared__ u16   sH[16 * 256];    // h (bf16), 8KB
  __shared__ float sC[16 * 256];    // c, 16KB
  __shared__ u16   sX[16 * 64];     // x tile (TDM dest), 2KB
  const int mblk = blockIdx.x, dir = blockIdx.y;
  const u16*   W    = dir ? Wcat_b : Wcat_f;
  const float* bias = dir ? bias_b : bias_f;
  const int tid = threadIdx.x, wave = tid >> 5, lane = tid & 31;
  const int arow = lane & 15, half = lane >> 4;
  for (int i = tid; i < 16 * 256; i += 1024) { sH[i] = 0; sC[i] = 0.f; }
  __syncthreads();
  const u16* hrow = &sH[arow * 256];
  const u16* xrow = &sX[arow * 64];
  const unsigned ldsa = (unsigned)(uintptr_t)&sX[0];   // LDS byte address (addr[31:0])
  for (int t = 0; t < 256; ++t) {
    const int tt = dir ? (255 - t) : t;
    // --- TDM: async 2D tile load (16 rows x 128B, row stride 32KB) into LDS ---
    if (wave == 0) {
      unsigned long long ga =
          (unsigned long long)(uintptr_t)(xbf + (((size_t)(mblk * 16) * 256 + tt) << 6));
      u32x4 g0 = { 1u,                                   // count=1 (valid user D#)
                   ldsa,                                  // lds_addr
                   (unsigned)ga,                          // global_addr[31:0]
                   (unsigned)((ga >> 32) & 0x1FFFFFFu) | (2u << 30) }; // addr[56:32]|type=2
      i32x8 g1 = { (int)0x10000u,        // data_size=1 (2 bytes)
                   (int)(64u << 16),     // tensor_dim0[15:0]=64 in bits[63:48]
                   (int)(16u << 16),     // tensor_dim1[15:0]=16 in bits[95:80]
                   (int)(64u << 16),     // tile_dim0=64 in bits[127:112]
                   (int)16u,             // tile_dim1=16 in bits[143:128]
                   (int)16384u,          // tensor_dim0_stride[31:0]=256*64 elements
                   0, 0 };
      i32x4 gz4 = { 0, 0, 0, 0 };
      i32x8 gz8 = { 0, 0, 0, 0, 0, 0, 0, 0 };
      __builtin_amdgcn_tensor_load_to_lds(g0, g1, gz4, gz4, gz8, 0);
      __builtin_amdgcn_s_wait_tensorcnt(0);
    }
    __syncthreads();   // x tile visible to all waves
#define ENC_A(k) ((k) < 64 ? *(const u16x8*)(xrow + (k)) : *(const u16x8*)(hrow + (k) - 64))
    for (int nt = wave; nt < 64; nt += 32) {
      const u16* wrow = W + (size_t)(nt * 16 + arow) * 320 + half * 8;
      v8f acc = (v8f)0.0f;
#pragma unroll
      for (int kk = 0; kk < 10; ++kk) {
        const int kb = kk * 32;
        u16x8 a0 = ENC_A(kb + half * 8);
        u16x8 a1 = ENC_A(kb + 16 + half * 8);
        u16x8 b0 = *(const u16x8*)(wrow + kb);
        u16x8 b1 = *(const u16x8*)(wrow + kb + 16);
        acc = wmma_bf16(a0, a1, b0, b1, acc);
      }
#pragma unroll
      for (int v = 0; v < 8; ++v)
        sG[(v + 8 * half) * 1024 + nt * 16 + arow] = acc[v];
    }
#undef ENC_A
    __syncthreads();
#pragma unroll
    for (int e = 0; e < 4; ++e) {
      int idx = tid + 1024 * e;                 // 0..4095
      int r = idx >> 8, j = idx & 255;
      float gi = sG[r * 1024 + j]       + bias[j];
      float gf = sG[r * 1024 + 256 + j] + bias[256 + j];
      float gg = sG[r * 1024 + 512 + j] + bias[512 + j];
      float go = sG[r * 1024 + 768 + j] + bias[768 + j];
      float c = sigf(gf) * sC[r * 256 + j] + sigf(gi) * tanhf(gg);
      float h = sigf(go) * tanhf(c);
      sC[r * 256 + j] = c;
      sH[r * 256 + j] = f2bf(h);
      ybf[(((size_t)(mblk * 16 + r) * 256 + tt) << 9) + dir * 256 + j] = f2bf(h);
    }
    __syncthreads();
  }
}

// ---------- encoder-out GEMM (131072 x 1024 x 512) + softplus epilogue ----------
__global__ __launch_bounds__(256) void k_encout(
    const u16* __restrict__ ybf, const u16* __restrict__ Wout,
    const float* __restrict__ bout, float* __restrict__ d_mu,
    float* __restrict__ d_lv) {
  const int mtile = blockIdx.x;                       // 0..8191
  const int tid = threadIdx.x, wave = tid >> 5, lane = tid & 31;
  const int arow = lane & 15, half = lane >> 4;
  const int nt = blockIdx.y * 8 + wave;               // 0..63
  const u16* ap   = ybf  + ((size_t)(mtile * 16 + arow) << 9);
  const u16* wrow = Wout + ((size_t)(nt * 16 + arow) << 9) + half * 8;
  __builtin_prefetch(ap + 256, 0, 0);                 // speculative: later K cachelines
  __builtin_prefetch(wrow + 256, 0, 0);
  v8f acc = (v8f)0.0f;
#pragma unroll
  for (int kk = 0; kk < 16; ++kk) {
    const int kb = kk * 32;
    u16x8 a0 = *(const u16x8*)(ap + kb + half * 8);
    u16x8 a1 = *(const u16x8*)(ap + kb + 16 + half * 8);
    u16x8 b0 = *(const u16x8*)(wrow + kb);
    u16x8 b1 = *(const u16x8*)(wrow + kb + 16);
    acc = wmma_bf16(a0, a1, b0, b1, acc);
  }
  const int n = nt * 16 + arow;
  const float bn = bout[n];
#pragma unroll
  for (int v = 0; v < 8; ++v) {
    int m = mtile * 16 + v + 8 * half;                // row = b*256+t
    float val = acc[v] + bn;
    if (n < 512) d_mu[(size_t)m * 512 + n] = val;
    else         d_lv[(size_t)m * 512 + (n - 512)] = softplusf(val);
  }
}

// ---------- latent: z = mu + eps * exp(2*log_var), store bf16 ----------
__global__ void k_zlatent(const float* __restrict__ mu, const float* __restrict__ lv,
                          const float* __restrict__ eps, u16* __restrict__ zbf) {
  size_t i = (size_t)blockIdx.x * 256 + threadIdx.x;  // B*T*512
  if (i >= (size_t)512 * 256 * 512) return;
  int n = (int)(i & 511); size_t r = i >> 9; int b = (int)(r >> 8);
  float z = mu[i] + eps[(size_t)b * 512 + n] * __expf(2.f * lv[i]);
  zbf[i] = f2bf(z);
}

// ---------- zdec GEMM (131072 x 32 x 512): z@Wz.T + bz -> d_z + bf16 (t<16) ----------
__global__ __launch_bounds__(64) void k_zdec(
    const u16* __restrict__ zbf, const u16* __restrict__ Wz,
    const float* __restrict__ bz, float* __restrict__ d_z,
    u16* __restrict__ zdec_bf) {
  const int mtile = blockIdx.x;                       // 0..8191
  const int tid = threadIdx.x, wave = tid >> 5, lane = tid & 31;
  const int arow = lane & 15, half = lane >> 4;
  const u16* ap   = zbf + ((size_t)(mtile * 16 + arow) << 9);
  const u16* wrow = Wz  + (size_t)(wave * 16 + arow) * 512 + half * 8;
  v8f acc = (v8f)0.0f;
#pragma unroll
  for (int kk = 0; kk < 16; ++kk) {
    const int kb = kk * 32;
    u16x8 a0 = *(const u16x8*)(ap + kb + half * 8);
    u16x8 a1 = *(const u16x8*)(ap + kb + 16 + half * 8);
    u16x8 b0 = *(const u16x8*)(wrow + kb);
    u16x8 b1 = *(const u16x8*)(wrow + kb + 16);
    acc = wmma_bf16(a0, a1, b0, b1, acc);
  }
  const int n = wave * 16 + arow;
  const float bn = bz[n];
#pragma unroll
  for (int v = 0; v < 8; ++v) {
    int m = mtile * 16 + v + 8 * half;                // row = b*256+t
    float val = acc[v] + bn;
    d_z[(size_t)m * 32 + n] = val;
    int t = m & 255, b = m >> 8;
    if (t < 16) zdec_bf[(size_t)((b << 4) + t) * 32 + n] = f2bf(val);
  }
}

// ---------- conductor LSTM: 16 steps, DEC=32, grid 32 ----------
__global__ __launch_bounds__(256) void k_conductor(
    const u16* __restrict__ zdec_bf, const u16* __restrict__ Wcon,
    const float* __restrict__ bcon, u16* __restrict__ emb_bf) {
  __shared__ float sG[16 * 128];
  __shared__ u16   sH[16 * 32];
  __shared__ float sC[16 * 32];
  const int mblk = blockIdx.x;
  const int tid = threadIdx.x, wave = tid >> 5, lane = tid & 31;
  const int arow = lane & 15, half = lane >> 4;
  for (int i = tid; i < 16 * 32; i += 256) { sH[i] = 0; sC[i] = 0.f; }
  __syncthreads();
  const u16* hrow = &sH[arow * 32];
  const u16* wrow = Wcon + (size_t)(wave * 16 + arow) * 64 + half * 8;
  for (int t = 0; t < 16; ++t) {
    const u16* ap = zdec_bf + (size_t)(((mblk * 16 + arow) << 4) + t) * 32;
#define CON_A(k) ((k) < 32 ? *(const u16x8*)(ap + (k)) : *(const u16x8*)(hrow + (k) - 32))
    v8f acc = (v8f)0.0f;
#pragma unroll
    for (int kk = 0; kk < 2; ++kk) {
      const int kb = kk * 32;
      u16x8 a0 = CON_A(kb + half * 8);
      u16x8 a1 = CON_A(kb + 16 + half * 8);
      u16x8 b0 = *(const u16x8*)(wrow + kb);
      u16x8 b1 = *(const u16x8*)(wrow + kb + 16);
      acc = wmma_bf16(a0, a1, b0, b1, acc);
    }
#undef CON_A
#pragma unroll
    for (int v = 0; v < 8; ++v)
      sG[(v + 8 * half) * 128 + wave * 16 + arow] = acc[v];
    __syncthreads();
#pragma unroll
    for (int e = 0; e < 2; ++e) {
      int idx = tid + 256 * e; int r = idx >> 5, j = idx & 31;
      float gi = sG[r * 128 + j]      + bcon[j];
      float gf = sG[r * 128 + 32 + j] + bcon[32 + j];
      float gg = sG[r * 128 + 64 + j] + bcon[64 + j];
      float go = sG[r * 128 + 96 + j] + bcon[96 + j];
      float c = sigf(gf) * sC[r * 32 + j] + sigf(gi) * tanhf(gg);
      float h = sigf(go) * tanhf(c);
      sC[r * 32 + j] = c; sH[r * 32 + j] = f2bf(h);
      emb_bf[(size_t)(((mblk * 16 + r) << 4) + t) * 32 + j] = f2bf(h);
    }
    __syncthreads();
  }
}

// ---------- decoder LSTM + output linear + softmax: grid 512 (8192 seqs) ----------
__global__ __launch_bounds__(256) void k_decoder(
    const u16* __restrict__ xbf, const u16* __restrict__ emb_bf,
    const u16* __restrict__ Wdec, const float* __restrict__ bdec,
    const u16* __restrict__ Wlin, const float* __restrict__ blin,
    const float* __restrict__ dec_h0, const float* __restrict__ dec_c0,
    float* __restrict__ d_notes) {
  __shared__ float sG[16 * 128];
  __shared__ u16   sH[16 * 32];
  __shared__ float sC[16 * 32];
  __shared__ float sL[16 * 64];
  const int sblk = blockIdx.x;                        // seq = sblk*16+r; b=seq>>4 tb=seq&15
  const int tid = threadIdx.x, wave = tid >> 5, lane = tid & 31;
  const int arow = lane & 15, half = lane >> 4;
  for (int i = tid; i < 16 * 32; i += 256) {
    int r = i >> 5, j = i & 31, seq = sblk * 16 + r, b = seq >> 4, tb = seq & 15;
    size_t off = ((size_t)tb * 512 + b) * 32 + j;     // dec_h0: (TB,B,DEC)
    sH[i] = f2bf(dec_h0[off]);
    sC[i] = dec_c0[off];
  }
  __syncthreads();
  const int seqA = sblk * 16 + arow, bA = seqA >> 4, tbA = seqA & 15;
  const u16* embrow = emb_bf + (size_t)((bA << 4) + tbA) * 32;
  const u16* hrow = &sH[arow * 32];
  const u16* wrow = Wdec + (size_t)(wave * 16 + arow) * 128 + half * 8;
  u16x8 zz = (u16x8)0;
  for (int s = 0; s < 16; ++s) {
    const int tgA = tbA * 16 + s;
    const bool tv = (tgA > 0);
    const u16* teach = xbf + ((size_t)(bA * 256 + (tgA - 1)) << 6);
#define DEC_A(k) ((k) < 32 ? *(const u16x8*)(embrow + (k)) : \
                  (k) < 96 ? (tv ? *(const u16x8*)(teach + (k) - 32) : zz) : \
                             *(const u16x8*)(hrow + (k) - 96))
    v8f acc = (v8f)0.0f;
#pragma unroll
    for (int kk = 0; kk < 4; ++kk) {
      const int kb = kk * 32;
      u16x8 a0 = DEC_A(kb + half * 8);
      u16x8 a1 = DEC_A(kb + 16 + half * 8);
      u16x8 b0 = *(const u16x8*)(wrow + kb);
      u16x8 b1 = *(const u16x8*)(wrow + kb + 16);
      acc = wmma_bf16(a0, a1, b0, b1, acc);
    }
#undef DEC_A
#pragma unroll
    for (int v = 0; v < 8; ++v)
      sG[(v + 8 * half) * 128 + wave * 16 + arow] = acc[v];
    __syncthreads();
#pragma unroll
    for (int e = 0; e < 2; ++e) {
      int idx = tid + 256 * e; int r = idx >> 5, j = idx & 31;
      float gi = sG[r * 128 + j]      + bdec[j];
      float gf = sG[r * 128 + 32 + j] + bdec[32 + j];
      float gg = sG[r * 128 + 64 + j] + bdec[64 + j];
      float go = sG[r * 128 + 96 + j] + bdec[96 + j];
      float c = sigf(gf) * sC[r * 32 + j] + sigf(gi) * tanhf(gg);
      float h = sigf(go) * tanhf(c);
      sC[r * 32 + j] = c; sH[r * 32 + j] = f2bf(h);
    }
    __syncthreads();
    if (wave < 4) {                                   // logits: 16x64 = h(16x32) @ Wlin.T
      const u16* wl = Wlin + (size_t)(wave * 16 + arow) * 32 + half * 8;
      u16x8 a0 = *(const u16x8*)(hrow + half * 8);
      u16x8 a1 = *(const u16x8*)(hrow + 16 + half * 8);
      u16x8 b0 = *(const u16x8*)(wl);
      u16x8 b1 = *(const u16x8*)(wl + 16);
      v8f acc2 = (v8f)0.0f;
      acc2 = wmma_bf16(a0, a1, b0, b1, acc2);
#pragma unroll
      for (int v = 0; v < 8; ++v)
        sL[(v + 8 * half) * 64 + wave * 16 + arow] = acc2[v];
    }
    __syncthreads();
    if (tid < 16) {                                   // softmax over 61 pitches
      int r = tid, seq = sblk * 16 + r, b = seq >> 4, tb = seq & 15;
      int tg = tb * 16 + s;
      float mx = -1e30f;
      for (int p = 0; p < 61; ++p) mx = fmaxf(mx, sL[r * 64 + p] + blin[p]);
      float sum = 0.f;
      for (int p = 0; p < 61; ++p) sum += __expf(sL[r * 64 + p] + blin[p] - mx);
      float inv = 1.f / sum;
      float* np = d_notes + ((size_t)b * 256 + tg) * 61;
      for (int p = 0; p < 61; ++p) np[p] = __expf(sL[r * 64 + p] + blin[p] - mx) * inv;
    }
    __syncthreads();
  }
}

// ---------- host ----------
extern "C" void kernel_launch(void* const* d_in, const int* in_sizes, int n_in,
                              void* d_out, int out_size, void* d_ws, size_t ws_size,
                              hipStream_t stream) {
  const float* x        = (const float*)d_in[0];
  const float* eps      = (const float*)d_in[1];
  const float* dec_h0   = (const float*)d_in[2];
  const float* dec_c0   = (const float*)d_in[3];
  const float* eWih_f   = (const float*)d_in[4];
  const float* eWhh_f   = (const float*)d_in[5];
  const float* eb_f     = (const float*)d_in[6];
  const float* eWih_b   = (const float*)d_in[7];
  const float* eWhh_b   = (const float*)d_in[8];
  const float* eb_b     = (const float*)d_in[9];
  const float* Wout     = (const float*)d_in[10];
  const float* bout     = (const float*)d_in[11];
  const float* Wz       = (const float*)d_in[12];
  const float* bz       = (const float*)d_in[13];
  const float* cWih     = (const float*)d_in[14];
  const float* cWhh     = (const float*)d_in[15];
  const float* cb       = (const float*)d_in[16];
  const float* dWih     = (const float*)d_in[17];
  const float* dWhh     = (const float*)d_in[18];
  const float* db       = (const float*)d_in[19];
  const float* Wlin     = (const float*)d_in[20];
  const float* blin     = (const float*)d_in[21];

  // output layout (f32): notes | z | mu | log_var
  float* out     = (float*)d_out;
  float* d_notes = out;
  float* d_z     = d_notes + (size_t)512 * 256 * 61;
  float* d_mu    = d_z     + (size_t)512 * 256 * 32;
  float* d_lv    = d_mu    + (size_t)512 * 256 * 512;

  // workspace bump allocator (all bf16 stored as u16)
  char* cur = (char*)d_ws;
  auto alloc = [&](size_t bytes) -> char* {
    char* p = cur; cur += (bytes + 255) & ~(size_t)255; return p;
  };
  u16* xbf     = (u16*)alloc((size_t)512 * 256 * 64 * 2);   // 16 MB
  u16* ybuf    = (u16*)alloc((size_t)512 * 256 * 512 * 2);  // 128 MB (y, reused as z_bf)
  u16* Wcat_f  = (u16*)alloc((size_t)1024 * 320 * 2);
  u16* Wcat_b  = (u16*)alloc((size_t)1024 * 320 * 2);
  u16* Wout_bf = (u16*)alloc((size_t)1024 * 512 * 2);
  u16* Wz_bf   = (u16*)alloc((size_t)32 * 512 * 2);
  u16* Wcon_bf = (u16*)alloc((size_t)128 * 64 * 2);
  u16* Wdec_bf = (u16*)alloc((size_t)128 * 128 * 2);
  u16* Wlin_bf = (u16*)alloc((size_t)64 * 32 * 2);
  u16* zdec_bf = (u16*)alloc((size_t)512 * 16 * 32 * 2);
  u16* emb_bf  = (u16*)alloc((size_t)512 * 16 * 32 * 2);
  (void)in_sizes; (void)n_in; (void)out_size; (void)ws_size;

  // pack / convert
  k_convert_x <<<(512 * 256 * 64 + 255) / 256, 256, 0, stream>>>(x, xbf);
  k_pack_enc  <<<(1024 * 320 + 255) / 256, 256, 0, stream>>>(eWih_f, eWhh_f, Wcat_f);
  k_pack_enc  <<<(1024 * 320 + 255) / 256, 256, 0, stream>>>(eWih_b, eWhh_b, Wcat_b);
  k_f32_to_bf16<<<(1024 * 512 + 255) / 256, 256, 0, stream>>>(Wout, Wout_bf, 1024 * 512);
  k_f32_to_bf16<<<(32 * 512 + 255) / 256, 256, 0, stream>>>(Wz, Wz_bf, 32 * 512);
  k_pack_con  <<<(128 * 64 + 255) / 256, 256, 0, stream>>>(cWih, cWhh, Wcon_bf);
  k_pack_dec  <<<(128 * 128 + 255) / 256, 256, 0, stream>>>(dWih, dWhh, Wdec_bf);
  k_pack_lin  <<<(64 * 32 + 255) / 256, 256, 0, stream>>>(Wlin, Wlin_bf);

  // model
  k_encoder   <<<dim3(32, 2), 1024, 0, stream>>>(xbf, Wcat_f, Wcat_b, eb_f, eb_b, ybuf);
  k_encout    <<<dim3(8192, 8), 256, 0, stream>>>(ybuf, Wout_bf, bout, d_mu, d_lv);
  k_zlatent   <<<262144, 256, 0, stream>>>(d_mu, d_lv, eps, ybuf);  // ybuf now holds z_bf
  k_zdec      <<<8192, 64, 0, stream>>>(ybuf, Wz_bf, bz, d_z, zdec_bf);
  k_conductor <<<32, 256, 0, stream>>>(zdec_bf, Wcon_bf, cb, emb_bf);
  k_decoder   <<<512, 256, 0, stream>>>(xbf, emb_bf, Wdec_bf, db, Wlin_bf, blin,
                                        dec_h0, dec_c0, d_notes);
}
